// HANEncoder_49280454754813
// MI455X (gfx1250) — compile-verified, matrix operand order checked by reading, not run on hardware
//
#include <hip/hip_runtime.h>
#include <cstdint>
#include <cstddef>

// ---------------- problem constants (match reference) ----------------
#define N_R_    50000
#define N_D_    10000
#define H_      8
#define Dh_     32
#define C_      256
#define RNA_IN_ 256
#define D_FEAT_ 512
#define E_RD_   200000
#define E_DR_   200000
#define E_RR_   400000
#define E_DD_   50000
#define NEG_SLOPE_ 0.2f

typedef __bf16 v16bf __attribute__((ext_vector_type(16)));
typedef float  v8f   __attribute__((ext_vector_type(8)));

// =====================================================================
// WMMA fragment layouts (per CDNA5 ISA 16-bit A 16x32 table):
//   lane l (both halves) holds row M = l&15
//   half kh = l>>4 : elems 0..7  = K = k0 + kh*8 + i
//                    elems 8..15 = K = k0 + 16 + kh*8 + i
// B (32x16 from [K,N] row-major weight): lane holds column N = l&15,
//   mirrored K mapping — pre-packed to fragment-major bf16 so a lane's
//   fragment is one contiguous 32B load.
// D (16x16 f32): lane l holds N = l&15, VGPR i holds M = (l>>4)*8 + i.
// =====================================================================

__device__ __forceinline__ v16bf frag_a_load(const float* __restrict__ A,
                                             int row, int ld, int k0, int kh) {
    const float* p = A + (size_t)row * ld + k0 + kh * 8;
    v16bf f;
#pragma unroll
    for (int i = 0; i < 8; ++i) {
        f[i]     = (__bf16)p[i];
        f[i + 8] = (__bf16)p[16 + i];
    }
    return f;
}

__device__ __forceinline__ v16bf frag_a2_load(const float* __restrict__ A0,
                                              const float* __restrict__ A1,
                                              float c0, float c1,
                                              int row, int ld, int k0, int kh) {
    size_t base = (size_t)row * ld + k0 + kh * 8;
    v16bf f;
#pragma unroll
    for (int i = 0; i < 8; ++i) {
        f[i]     = (__bf16)(c0 * A0[base + i]      + c1 * A1[base + i]);
        f[i + 8] = (__bf16)(c0 * A0[base + 16 + i] + c1 * A1[base + 16 + i]);
    }
    return f;
}

// ---- pre-pack weight [K,N] f32 -> fragment-major bf16 ----
// frag index t = nt*(K/32)+ks ; element layout matches B-fragment of lane.
__global__ __launch_bounds__(256)
void pack_w_kernel(const float* __restrict__ W, __bf16* __restrict__ out,
                   int K, int N) {
    int idx = blockIdx.x * blockDim.x + threadIdx.x;
    int ksteps = K >> 5;
    int total = (N >> 4) * ksteps * 32;
    if (idx >= total) return;
    int lane = idx & 31;
    int t = idx >> 5;
    int ks = t % ksteps, nt = t / ksteps;
    int kh = lane >> 4, l15 = lane & 15;
    const float* p = W + (size_t)(ks * 32 + kh * 8) * N + nt * 16 + l15;
    __bf16* o = out + (size_t)idx * 16;
#pragma unroll
    for (int i = 0; i < 8; ++i) {
        o[i]     = (__bf16)p[(size_t)i * N];
        o[i + 8] = (__bf16)p[(size_t)(16 + i) * N];
    }
}

// ---------------- generic tile GEMM (used for K=512 projection) ----------------
__global__ __launch_bounds__(256)
void wmma_gemm_bias_packed(const float* __restrict__ A, const __bf16* __restrict__ Wp,
                           const float* __restrict__ bias, float* __restrict__ Cm,
                           int M, int K, int N) {
    int wid  = (blockIdx.x * blockDim.x + threadIdx.x) >> 5;
    int lane = threadIdx.x & 31;
    int tilesN = N >> 4, ksteps = K >> 5;
    int total = (M >> 4) * tilesN;
    if (wid >= total) return;
    int tm = wid / tilesN, tn = wid % tilesN;
    int row0 = tm << 4, col0 = tn << 4;
    int kh = lane >> 4, l15 = lane & 15;
    int arow = row0 + l15;
    const v16bf* bp = (const v16bf*)Wp + (size_t)tn * ksteps * 32 + lane;

    v8f acc = {};
    for (int ks = 0; ks < ksteps; ++ks) {
        v16bf af = frag_a_load(A, arow, K, ks * 32, kh);
        v16bf bf = bp[(size_t)ks * 32];
        acc = __builtin_amdgcn_wmma_f32_16x16x32_bf16(false, af, false, bf,
                                                      (short)0, acc, false, false);
    }
    float bv = bias[col0 + l15];
#pragma unroll
    for (int i = 0; i < 8; ++i)
        Cm[(size_t)(row0 + kh * 8 + i) * N + col0 + l15] = acc[i] + bv;
}

// ---------------- K=N=256 row-panel GEMM: A cached in registers ----------------
__global__ __launch_bounds__(256)
void wmma_gemm256_bias(const float* __restrict__ A, const __bf16* __restrict__ Wp,
                       const float* __restrict__ bias, float* __restrict__ Cm, int M) {
    int wid  = (blockIdx.x * blockDim.x + threadIdx.x) >> 5;
    int lane = threadIdx.x & 31;
    if (wid >= (M >> 4)) return;
    int row0 = wid << 4;
    int kh = lane >> 4, l15 = lane & 15;
    int arow = row0 + l15;

    v16bf a[8];
#pragma unroll
    for (int ks = 0; ks < 8; ++ks) a[ks] = frag_a_load(A, arow, 256, ks * 32, kh);

    const v16bf* bp = (const v16bf*)Wp + lane;
    for (int nt = 0; nt < 16; ++nt) {
        v8f acc = {};
#pragma unroll
        for (int ks = 0; ks < 8; ++ks) {
            v16bf bf = bp[(size_t)(nt * 8 + ks) * 32];
            acc = __builtin_amdgcn_wmma_f32_16x16x32_bf16(false, a[ks], false, bf,
                                                          (short)0, acc, false, false);
        }
        int col0 = nt << 4;
        float bv = bias[col0 + l15];
#pragma unroll
        for (int i = 0; i < 8; ++i)
            Cm[(size_t)(row0 + kh * 8 + i) * C_ + col0 + l15] = acc[i] + bv;
    }
}

// ---- score += sum_rows q . tanh(A @ kW + kb)  (semantic attention, fused) ----
__global__ __launch_bounds__(256)
void wmma_gemm256_tanh_qdot(const float* __restrict__ A, const __bf16* __restrict__ Wp,
                            const float* __restrict__ bias, const float* __restrict__ qv,
                            float* __restrict__ scoreOut, int M) {
    int wid  = (blockIdx.x * blockDim.x + threadIdx.x) >> 5;
    int lane = threadIdx.x & 31;
    if (wid >= (M >> 4)) return;
    int row0 = wid << 4;
    int kh = lane >> 4, l15 = lane & 15;
    int arow = row0 + l15;

    v16bf a[8];
#pragma unroll
    for (int ks = 0; ks < 8; ++ks) a[ks] = frag_a_load(A, arow, 256, ks * 32, kh);

    const v16bf* bp = (const v16bf*)Wp + lane;
    float part = 0.f;
    for (int nt = 0; nt < 16; ++nt) {
        v8f acc = {};
#pragma unroll
        for (int ks = 0; ks < 8; ++ks) {
            v16bf bf = bp[(size_t)(nt * 8 + ks) * 32];
            acc = __builtin_amdgcn_wmma_f32_16x16x32_bf16(false, a[ks], false, bf,
                                                          (short)0, acc, false, false);
        }
        int col0 = nt << 4;
        float bv = bias[col0 + l15];
        float qc = qv[col0 + l15];
#pragma unroll
        for (int i = 0; i < 8; ++i) part += qc * tanhf(acc[i] + bv);
    }
    // wave32 reduction, one atomic per wave
#pragma unroll
    for (int off = 16; off > 0; off >>= 1) part += __shfl_down(part, off);
    if (lane == 0) atomicAdd(scoreOut, part);
}

// ---- out = (c0*A0 + c1*A1) @ W + bias  (semantic combine fused, A built once) ----
__global__ __launch_bounds__(256)
void wmma_gemm256_combined(const float* __restrict__ A0, const float* __restrict__ A1,
                           const float* __restrict__ attn2, const __bf16* __restrict__ Wp,
                           const float* __restrict__ bias, float* __restrict__ Cm, int M) {
    int wid  = (blockIdx.x * blockDim.x + threadIdx.x) >> 5;
    int lane = threadIdx.x & 31;
    if (wid >= (M >> 4)) return;
    int row0 = wid << 4;
    int kh = lane >> 4, l15 = lane & 15;
    int arow = row0 + l15;
    float c0 = attn2[0], c1 = attn2[1];

    v16bf a[8];
#pragma unroll
    for (int ks = 0; ks < 8; ++ks)
        a[ks] = frag_a2_load(A0, A1, c0, c1, arow, 256, ks * 32, kh);

    const v16bf* bp = (const v16bf*)Wp + lane;
    for (int nt = 0; nt < 16; ++nt) {
        v8f acc = {};
#pragma unroll
        for (int ks = 0; ks < 8; ++ks) {
            v16bf bf = bp[(size_t)(nt * 8 + ks) * 32];
            acc = __builtin_amdgcn_wmma_f32_16x16x32_bf16(false, a[ks], false, bf,
                                                          (short)0, acc, false, false);
        }
        int col0 = nt << 4;
        float bv = bias[col0 + l15];
#pragma unroll
        for (int i = 0; i < 8; ++i)
            Cm[(size_t)(row0 + kh * 8 + i) * C_ + col0 + l15] = acc[i] + bv;
    }
}

// =====================================================================
// Edge-attention (memory/atomic-bound scatter) kernels
// =====================================================================

__device__ __forceinline__ unsigned fenc(float f) {
    unsigned u = __float_as_uint(f);
    return (f >= 0.f) ? (u | 0x80000000u) : ~u;
}
__device__ __forceinline__ float fdec(unsigned k) {
    if (k == 0u) return 0.f;  // empty segment -> where(isfinite(m), m, 0)
    return (k & 0x80000000u) ? __uint_as_float(k & 0x7FFFFFFFu) : __uint_as_float(~k);
}

__global__ __launch_bounds__(256)
void node_score_kernel(const float* __restrict__ x, const float* __restrict__ a,
                       float* __restrict__ s, int n) {
    int idx = blockIdx.x * blockDim.x + threadIdx.x;
    if (idx >= n * H_) return;
    int node = idx >> 3, h = idx & 7;
    const float* xp = x + (size_t)node * C_ + h * Dh_;
    const float* ap = a + h * Dh_;
    float acc = 0.f;
#pragma unroll
    for (int d = 0; d < Dh_; ++d) acc += xp[d] * ap[d];
    s[idx] = acc;
}

__device__ __forceinline__ float edge_alpha(const float* ss, const float* sd,
                                            int s, int d, int h) {
    float al = ss[s * H_ + h] + sd[d * H_ + h];
    return al > 0.f ? al : NEG_SLOPE_ * al;
}

__global__ __launch_bounds__(256)
void edge_max_kernel(const float* __restrict__ ss, const float* __restrict__ sd,
                     const int* __restrict__ src, const int* __restrict__ dst,
                     unsigned* __restrict__ mk, int nE) {
    int idx = blockIdx.x * blockDim.x + threadIdx.x;
    if (idx >= nE * H_) return;
    int e = idx >> 3, h = idx & 7;
    int d = dst[e];
    float alpha = edge_alpha(ss, sd, src[e], d, h);
    atomicMax(&mk[d * H_ + h], fenc(alpha));
}

__global__ __launch_bounds__(256)
void edge_sum_kernel(const float* __restrict__ ss, const float* __restrict__ sd,
                     const int* __restrict__ src, const int* __restrict__ dst,
                     const unsigned* __restrict__ mk, float* __restrict__ dn, int nE) {
    int idx = blockIdx.x * blockDim.x + threadIdx.x;
    if (idx >= nE * H_) return;
    int e = idx >> 3, h = idx & 7;
    int d = dst[e];
    float alpha = edge_alpha(ss, sd, src[e], d, h);
    float m = fdec(mk[d * H_ + h]);
    atomicAdd(&dn[d * H_ + h], expf(alpha - m));
}

__global__ __launch_bounds__(256)
void edge_agg_kernel(const float* __restrict__ xsrc,
                     const float* __restrict__ ss, const float* __restrict__ sd,
                     const int* __restrict__ src, const int* __restrict__ dst,
                     const unsigned* __restrict__ mk, const float* __restrict__ dn,
                     float* __restrict__ outp, int nE) {
    int idx = blockIdx.x * blockDim.x + threadIdx.x;
    if (idx >= nE * H_) return;
    int e = idx >> 3, h = idx & 7;
    int s = src[e], d = dst[e];
    float alpha = edge_alpha(ss, sd, s, d, h);
    float m = fdec(mk[d * H_ + h]);
    float w = expf(alpha - m) / (dn[d * H_ + h] + 1e-16f);
    const float* xp = xsrc + (size_t)s * C_ + h * Dh_;
    float* op = outp + (size_t)d * C_ + h * Dh_;
    __builtin_prefetch(xp, 0, 0);
#pragma unroll
    for (int k = 0; k < Dh_; ++k) atomicAdd(&op[k], xp[k] * w);
}

__global__ __launch_bounds__(256)
void relu_kernel(float* __restrict__ x, int n) {
    int idx = blockIdx.x * blockDim.x + threadIdx.x;
    if (idx < n) x[idx] = fmaxf(x[idx], 0.f);
}

__global__ void softmax_pairs_kernel(const float* __restrict__ score,
                                     float* __restrict__ attn,
                                     float invR, float invD) {
    if (blockIdx.x == 0 && threadIdx.x == 0) {
        float s0 = score[0] * invR, s1 = score[1] * invR;
        float m = fmaxf(s0, s1);
        float e0 = expf(s0 - m), e1 = expf(s1 - m);
        attn[0] = e0 / (e0 + e1); attn[1] = e1 / (e0 + e1);
        s0 = score[2] * invD; s1 = score[3] * invD;
        m = fmaxf(s0, s1);
        e0 = expf(s0 - m); e1 = expf(s1 - m);
        attn[2] = e0 / (e0 + e1); attn[3] = e1 / (e0 + e1);
    }
}

// =====================================================================
// Host launcher
// =====================================================================
extern "C" void kernel_launch(void* const* d_in, const int* in_sizes, int n_in,
                              void* d_out, int out_size, void* d_ws, size_t ws_size,
                              hipStream_t stream) {
    (void)in_sizes; (void)n_in; (void)out_size; (void)ws_size;

    const float* x_rna    = (const float*)d_in[0];
    const float* x_dis    = (const float*)d_in[1];
    const float* W_d      = (const float*)d_in[2];
    const float* b_d      = (const float*)d_in[3];
    const float* Wp_rna   = (const float*)d_in[4];
    const float* bp_rna   = (const float*)d_in[5];
    const float* Wp_dis   = (const float*)d_in[6];
    const float* bp_dis   = (const float*)d_in[7];
    const float* a_src_rd = (const float*)d_in[8];
    const float* a_dst_rd = (const float*)d_in[9];
    const float* a_src_dr = (const float*)d_in[10];
    const float* a_dst_dr = (const float*)d_in[11];
    const float* a_src_rr = (const float*)d_in[12];
    const float* a_dst_rr = (const float*)d_in[13];
    const float* a_src_dd = (const float*)d_in[14];
    const float* a_dst_dd = (const float*)d_in[15];
    const float* kW       = (const float*)d_in[16];
    const float* kb       = (const float*)d_in[17];
    const float* qv       = (const float*)d_in[18];
    const float* W_out    = (const float*)d_in[19];
    const float* b_out    = (const float*)d_in[20];
    const int* ei_rd_src  = (const int*)d_in[21];
    const int* ei_rd_dst  = (const int*)d_in[22];
    const int* ei_dr_src  = (const int*)d_in[23];
    const int* ei_dr_dst  = (const int*)d_in[24];
    const int* ei_rr_src  = (const int*)d_in[25];
    const int* ei_rr_dst  = (const int*)d_in[26];
    const int* ei_dd_src  = (const int*)d_in[27];
    const int* ei_dd_dst  = (const int*)d_in[28];
    float* dout = (float*)d_out;

    // ---- workspace bump allocator (256B aligned chunks) ----
    char* wsb = (char*)d_ws;
    size_t off = 0;
    auto alloc = [&](size_t nbytes) -> void* {
        void* p = wsb + off;
        off += (nbytes + 255) & ~(size_t)255;
        return p;
    };
    float* xd = (float*)alloc((size_t)N_D_ * C_ * 4);
    float* hr = (float*)alloc((size_t)N_R_ * C_ * 4);
    float* hd = (float*)alloc((size_t)N_D_ * C_ * 4);
    float* s_rd_s = (float*)alloc((size_t)N_R_ * H_ * 4);
    float* s_rd_d = (float*)alloc((size_t)N_D_ * H_ * 4);
    float* s_dr_s = (float*)alloc((size_t)N_D_ * H_ * 4);
    float* s_dr_d = (float*)alloc((size_t)N_R_ * H_ * 4);
    float* s_rr_s = (float*)alloc((size_t)N_R_ * H_ * 4);
    float* s_rr_d = (float*)alloc((size_t)N_R_ * H_ * 4);
    float* s_dd_s = (float*)alloc((size_t)N_D_ * H_ * 4);
    float* s_dd_d = (float*)alloc((size_t)N_D_ * H_ * 4);
    // packed bf16 weights (fully overwritten each call by pack kernels)
    __bf16* Wd_p   = (__bf16*)alloc((size_t)D_FEAT_ * C_ * 2);
    __bf16* Wpr_p  = (__bf16*)alloc((size_t)RNA_IN_ * C_ * 2);
    __bf16* Wpd_p  = (__bf16*)alloc((size_t)RNA_IN_ * C_ * 2);
    __bf16* kW_p   = (__bf16*)alloc((size_t)C_ * C_ * 2);
    __bf16* Wout_p = (__bf16*)alloc((size_t)C_ * C_ * 2);
    size_t zero_begin = off;   // everything below zeroed each call
    unsigned* mk_rd = (unsigned*)alloc((size_t)N_D_ * H_ * 4);
    unsigned* mk_dr = (unsigned*)alloc((size_t)N_R_ * H_ * 4);
    unsigned* mk_rr = (unsigned*)alloc((size_t)N_R_ * H_ * 4);
    unsigned* mk_dd = (unsigned*)alloc((size_t)N_D_ * H_ * 4);
    float* dn_rd = (float*)alloc((size_t)N_D_ * H_ * 4);
    float* dn_dr = (float*)alloc((size_t)N_R_ * H_ * 4);
    float* dn_rr = (float*)alloc((size_t)N_R_ * H_ * 4);
    float* dn_dd = (float*)alloc((size_t)N_D_ * H_ * 4);
    float* out_rd = (float*)alloc((size_t)N_D_ * C_ * 4);
    float* out_dr = (float*)alloc((size_t)N_R_ * C_ * 4);
    float* out_rr = (float*)alloc((size_t)N_R_ * C_ * 4);
    float* out_dd = (float*)alloc((size_t)N_D_ * C_ * 4);
    float* score = (float*)alloc(4 * 4);
    float* attn  = (float*)alloc(4 * 4);
    size_t zero_end = off;

    hipMemsetAsync(wsb + zero_begin, 0, zero_end - zero_begin, stream);

    auto cdiv = [](long a, long b) { return (int)((a + b - 1) / b); };
    dim3 blk(256);
    auto pgrid = [&](int K) { return dim3(cdiv((long)(C_ / 16) * (K / 32) * 32, 256)); };
    auto rgrid = [&](int M) { return dim3(cdiv(M / 16, 8)); };  // row-panel kernels

    // ---- pack weights to fragment-major bf16 ----
    pack_w_kernel<<<pgrid(D_FEAT_), blk, 0, stream>>>(W_d, Wd_p, D_FEAT_, C_);
    pack_w_kernel<<<pgrid(RNA_IN_), blk, 0, stream>>>(Wp_rna, Wpr_p, RNA_IN_, C_);
    pack_w_kernel<<<pgrid(RNA_IN_), blk, 0, stream>>>(Wp_dis, Wpd_p, RNA_IN_, C_);
    pack_w_kernel<<<pgrid(C_), blk, 0, stream>>>(kW, kW_p, C_, C_);
    pack_w_kernel<<<pgrid(C_), blk, 0, stream>>>(W_out, Wout_p, C_, C_);

    // ---- projections ----
    wmma_gemm_bias_packed<<<dim3(cdiv((long)(N_D_ / 16) * (C_ / 16), 8)), blk, 0, stream>>>(
        x_dis, Wd_p, b_d, xd, N_D_, D_FEAT_, C_);
    wmma_gemm256_bias<<<rgrid(N_R_), blk, 0, stream>>>(x_rna, Wpr_p, bp_rna, hr, N_R_);
    wmma_gemm256_bias<<<rgrid(N_D_), blk, 0, stream>>>(xd, Wpd_p, bp_dis, hd, N_D_);

    // ---- per-node attention scores ----
    auto ns = [&](const float* x, const float* a, float* s, int n) {
        node_score_kernel<<<cdiv((long)n * H_, 256), blk, 0, stream>>>(x, a, s, n);
    };
    ns(hr, a_src_rd, s_rd_s, N_R_); ns(hd, a_dst_rd, s_rd_d, N_D_);
    ns(hd, a_src_dr, s_dr_s, N_D_); ns(hr, a_dst_dr, s_dr_d, N_R_);
    ns(hr, a_src_rr, s_rr_s, N_R_); ns(hr, a_dst_rr, s_rr_d, N_R_);
    ns(hd, a_src_dd, s_dd_s, N_D_); ns(hd, a_dst_dd, s_dd_d, N_D_);

    // ---- edge softmax + aggregation (3 passes per edge type) ----
    auto run_edges = [&](const float* xs, const float* ss, const float* sd,
                         const int* es, const int* ed, unsigned* mk, float* dn,
                         float* op, int nE) {
        dim3 g(cdiv((long)nE * H_, 256));
        edge_max_kernel<<<g, blk, 0, stream>>>(ss, sd, es, ed, mk, nE);
        edge_sum_kernel<<<g, blk, 0, stream>>>(ss, sd, es, ed, mk, dn, nE);
        edge_agg_kernel<<<g, blk, 0, stream>>>(xs, ss, sd, es, ed, mk, dn, op, nE);
    };
    run_edges(hr, s_rd_s, s_rd_d, ei_rd_src, ei_rd_dst, mk_rd, dn_rd, out_rd, E_RD_);
    run_edges(hd, s_dr_s, s_dr_d, ei_dr_src, ei_dr_dst, mk_dr, dn_dr, out_dr, E_DR_);
    run_edges(hr, s_rr_s, s_rr_d, ei_rr_src, ei_rr_dst, mk_rr, dn_rr, out_rr, E_RR_);
    run_edges(hd, s_dd_s, s_dd_d, ei_dd_src, ei_dd_dst, mk_dd, dn_dd, out_dd, E_DD_);

    relu_kernel<<<cdiv((long)N_D_ * C_, 256), blk, 0, stream>>>(out_rd, N_D_ * C_);
    relu_kernel<<<cdiv((long)N_R_ * C_, 256), blk, 0, stream>>>(out_dr, N_R_ * C_);
    relu_kernel<<<cdiv((long)N_R_ * C_, 256), blk, 0, stream>>>(out_rr, N_R_ * C_);
    relu_kernel<<<cdiv((long)N_D_ * C_, 256), blk, 0, stream>>>(out_dd, N_D_ * C_);

    // ---- semantic attention scores (fused GEMM+tanh+q-dot) ----
    wmma_gemm256_tanh_qdot<<<rgrid(N_R_), blk, 0, stream>>>(out_dr, kW_p, kb, qv, score + 0, N_R_);
    wmma_gemm256_tanh_qdot<<<rgrid(N_R_), blk, 0, stream>>>(out_rr, kW_p, kb, qv, score + 1, N_R_);
    wmma_gemm256_tanh_qdot<<<rgrid(N_D_), blk, 0, stream>>>(out_rd, kW_p, kb, qv, score + 2, N_D_);
    wmma_gemm256_tanh_qdot<<<rgrid(N_D_), blk, 0, stream>>>(out_dd, kW_p, kb, qv, score + 3, N_D_);

    softmax_pairs_kernel<<<1, 1, 0, stream>>>(score, attn, 1.f / N_R_, 1.f / N_D_);

    // ---- final linear with fused semantic combination ----
    wmma_gemm256_combined<<<rgrid(N_R_), blk, 0, stream>>>(out_dr, out_rr, attn + 0, Wout_p,
                                                           b_out, dout, N_R_);
    wmma_gemm256_combined<<<rgrid(N_D_), blk, 0, stream>>>(out_rd, out_dd, attn + 2, Wout_p,
                                                           b_out, dout + (size_t)N_R_ * C_, N_D_);
}
